// HierTextBranch_80367428043347
// MI455X (gfx1250) — compile-verified, matrix operand order checked by reading, not run on hardware
//
#include <hip/hip_runtime.h>

// ---------------------------------------------------------------------------
// Problem constants (from the reference)
#define N_     4096
#define D_     512
#define HEADS_ 8
#define DH_    64
#define HID_   256
#define S_     64

typedef __attribute__((ext_vector_type(16))) _Float16 v16h;
typedef __attribute__((ext_vector_type(8)))  float    v8f;
typedef __attribute__((ext_vector_type(4)))  unsigned int u4;
typedef __attribute__((ext_vector_type(4)))  int      int4v;

union V16 { v16h v; u4 u[2]; };

static __device__ inline v8f wmma16(v16h a, v16h b, v8f c) {
  // D(f32 16x16) = A(f16 16x32) * B(f16 32x16) + C
  return __builtin_amdgcn_wmma_f32_16x16x32_f16(false, a, false, b, (short)0, c,
                                                false, false);
}

// ---- CDNA5 async global->LDS path (guarded; falls back to sync copies) ----
#if __has_builtin(__builtin_amdgcn_global_load_async_to_lds_b128)
#define HAVE_ASYNC_LDS 1
static __device__ inline void async_cp16(const _Float16* g, _Float16* l) {
  __builtin_amdgcn_global_load_async_to_lds_b128(
      (__attribute__((address_space(1))) int4v*)(unsigned long long)g,
      (__attribute__((address_space(3))) int4v*)(unsigned long long)l,  // low 32b = LDS offset
      0, 0);
}
#if __has_builtin(__builtin_amdgcn_s_wait_asynccnt)
#define WAIT_ASYNC(n) __builtin_amdgcn_s_wait_asynccnt(n)
#else
#define WAIT_ASYNC(n) asm volatile("s_wait_asynccnt %0" ::"i"(n) : "memory")
#endif
#else
#define HAVE_ASYNC_LDS 0
#define WAIT_ASYNC(n)
#endif

// Block-wide reduction (blockDim.x == 256).  mode: 0=sum, 1=max.
static __device__ float block_reduce(float v, int mode) {
  __shared__ float buf[8];
  const int lane = threadIdx.x & 31;
  const int wid  = threadIdx.x >> 5;
  #pragma unroll
  for (int off = 16; off > 0; off >>= 1) {
    float o = __shfl_xor(v, off, 32);
    v = mode ? fmaxf(v, o) : (v + o);
  }
  __syncthreads();
  if (lane == 0) buf[wid] = v;
  __syncthreads();
  float r = mode ? -3.0e38f : 0.0f;
  #pragma unroll
  for (int i = 0; i < 8; ++i) r = mode ? fmaxf(r, buf[i]) : (r + buf[i]);
  return r;
}

// ---------------------------------------------------------------------------
// Prep kernels
__global__ void cvt_f32_f16(const float* __restrict__ a, _Float16* __restrict__ b, int n) {
  int i = blockIdx.x * 256 + threadIdx.x;
  if (i < n) b[i] = (_Float16)a[i];
}

// B [K, Nc] f32  ->  Bt [Nc, K] f16
__global__ void trans_f32_f16(const float* __restrict__ B, _Float16* __restrict__ Bt,
                              int K, int Nc) {
  int i = blockIdx.x * 256 + threadIdx.x;
  if (i < K * Nc) {
    int n = i / K, k = i % K;
    Bt[i] = (_Float16)B[(size_t)k * Nc + n];
  }
}

// A [R, C] f16 -> At [C, R] f16
__global__ void trans_f16(const _Float16* __restrict__ A, _Float16* __restrict__ At,
                          int R, int C) {
  int i = blockIdx.x * 256 + threadIdx.x;
  if (i < R * C) {
    int c = i / R, r = i % R;
    At[i] = A[(size_t)r * C + c];
  }
}

// ---------------------------------------------------------------------------
// Generic WMMA GEMM: out = act(A[M,K] * B[K,Nc] + bias), with B given as
// Bt[Nc,K] (row-major transpose).  M%128==0, Nc%64==0, K%64==0.
// act: 0 = identity, 1 = exact GELU, 2 = tanh.
// Double-buffered LDS, 64-deep K stages, async global->LDS when available.
#define KT 64
#define LDSW 72  // halves per row; 144B rows keep all 16B chunks aligned
__global__ __launch_bounds__(256) void gemm_ws(const _Float16* __restrict__ A,
                                               const _Float16* __restrict__ Bt,
                                               const float* __restrict__ bias,
                                               float* __restrict__ outF,
                                               _Float16* __restrict__ outH,
                                               int M, int Nc, int K, int act) {
  __shared__ _Float16 As[2][128 * LDSW];
  __shared__ _Float16 Bs[2][64 * LDSW];
  const int tid  = threadIdx.x;
  const int lane = tid & 31, wave = tid >> 5;
  const int lg = lane >> 4, ln = lane & 15;
  const int wm = wave >> 1, wn = wave & 1;          // 4 x 2 wave grid
  const int rowBase = blockIdx.y * 128;
  const int colBase = blockIdx.x * 64;
  v8f acc[2][2] = {};

  // Stage loader: A tile 128x64 (1024 16B-chunks -> 4/thread),
  //               B tile  64x64 ( 512 16B-chunks -> 2/thread).
  auto load_stage = [&](int buf, int k0) {
    #pragma unroll
    for (int i = 0; i < 4; ++i) {
      int chunk = tid + i * 256;
      int row = chunk >> 3, col = (chunk & 7) * 8;
      const _Float16* src = A + (size_t)(rowBase + row) * K + k0 + col;
      _Float16* dst = &As[buf][row * LDSW + col];
#if HAVE_ASYNC_LDS
      async_cp16(src, dst);
#else
      *(u4*)dst = *(const u4*)src;
#endif
    }
    #pragma unroll
    for (int i = 0; i < 2; ++i) {
      int chunk = tid + i * 256;
      int row = chunk >> 3, col = (chunk & 7) * 8;
      const _Float16* src = Bt + (size_t)(colBase + row) * K + k0 + col;
      _Float16* dst = &Bs[buf][row * LDSW + col];
#if HAVE_ASYNC_LDS
      async_cp16(src, dst);
#else
      *(u4*)dst = *(const u4*)src;
#endif
    }
  };

  const int nst = K / KT;
  load_stage(0, 0);
  for (int i = 0; i < nst; ++i) {
    const int buf = i & 1;
    if (i + 1 < nst) {
      load_stage(buf ^ 1, (i + 1) * KT);
      WAIT_ASYNC(6);   // oldest stage (6 async ops/thread) complete
    } else {
      WAIT_ASYNC(0);
    }
    __syncthreads();

    #pragma unroll
    for (int kk = 0; kk < KT; kk += 32) {
      V16 af[2], bf[2];
      #pragma unroll
      for (int mt = 0; mt < 2; ++mt) {
        const _Float16* s = &As[buf][(wm * 32 + mt * 16 + ln) * LDSW + kk + lg * 8];
        af[mt].u[0] = *(const u4*)s;
        af[mt].u[1] = *(const u4*)(s + 16);
      }
      #pragma unroll
      for (int nt = 0; nt < 2; ++nt) {
        const _Float16* s = &Bs[buf][(wn * 32 + nt * 16 + ln) * LDSW + kk + lg * 8];
        bf[nt].u[0] = *(const u4*)s;
        bf[nt].u[1] = *(const u4*)(s + 16);
      }
      #pragma unroll
      for (int mt = 0; mt < 2; ++mt)
        #pragma unroll
        for (int nt = 0; nt < 2; ++nt)
          acc[mt][nt] = wmma16(af[mt].v, bf[nt].v, acc[mt][nt]);
    }
    __syncthreads();   // reads done before this buffer is refilled
  }

  #pragma unroll
  for (int mt = 0; mt < 2; ++mt)
    #pragma unroll
    for (int nt = 0; nt < 2; ++nt)
      #pragma unroll
      for (int r = 0; r < 8; ++r) {
        int m = rowBase + wm * 32 + mt * 16 + lg * 8 + r;
        int c = colBase + wn * 32 + nt * 16 + ln;
        float v = acc[mt][nt][r];
        if (bias) v += bias[c];
        if (act == 1) v = 0.5f * v * (1.0f + erff(v * 0.70710678118f));
        else if (act == 2) v = tanhf(v);
        size_t idx = (size_t)m * Nc + c;
        if (outF) outF[idx] = v;
        if (outH) outH[idx] = (_Float16)v;
      }
}

// ---------------------------------------------------------------------------
// Flash attention.  grid = (N/64, HEADS), block = 128 (4 waves).
// Q,K: f16 [N, D] (head h occupies cols h*64..h*64+63).  Vt: f16 [D, N]
// (row h*64+d holds V[:, h, d]).  Output O: f16 [N, D].
__global__ __launch_bounds__(128) void fa_ws(const _Float16* __restrict__ Q,
                                             const _Float16* __restrict__ Kf,
                                             const _Float16* __restrict__ Vt,
                                             _Float16* __restrict__ O) {
  __shared__ _Float16 pbuf[4][16][72];   // wave-private P tiles (16x64, padded)
  const int h    = blockIdx.y;
  const int wave = threadIdx.x >> 5;
  const int lane = threadIdx.x & 31;
  const int lg = lane >> 4, ln = lane & 15;
  const int qrow = blockIdx.x * 64 + wave * 16;
  const float scale = 0.125f;            // 1/sqrt(DH)

  V16 qf[2];
  {
    const _Float16* qp = Q + (size_t)(qrow + ln) * D_ + h * DH_;
    #pragma unroll
    for (int kc = 0; kc < 2; ++kc) {
      qf[kc].u[0] = *(const u4*)(qp + kc * 32 + lg * 8);
      qf[kc].u[1] = *(const u4*)(qp + kc * 32 + lg * 8 + 16);
    }
  }

  v8f o[4] = {};
  float mrow[8], lrow[8];
  #pragma unroll
  for (int r = 0; r < 8; ++r) { mrow[r] = -3.0e38f; lrow[r] = 0.0f; }

  for (int jb = 0; jb < N_; jb += 64) {
    // S = Q * K^T for 64 keys
    v8f s[4] = {};
    #pragma unroll
    for (int nt = 0; nt < 4; ++nt) {
      const _Float16* kp = Kf + (size_t)(jb + nt * 16 + ln) * D_ + h * DH_;
      #pragma unroll
      for (int kc = 0; kc < 2; ++kc) {
        V16 b;
        b.u[0] = *(const u4*)(kp + kc * 32 + lg * 8);
        b.u[1] = *(const u4*)(kp + kc * 32 + lg * 8 + 16);
        s[nt] = wmma16(qf[kc].v, b.v, s[nt]);
      }
    }
    // online softmax (row = lg*8 + r, cols striped over 16 lanes x 4 tiles)
    #pragma unroll
    for (int r = 0; r < 8; ++r) {
      float mx = fmaxf(fmaxf(s[0][r], s[1][r]), fmaxf(s[2][r], s[3][r])) * scale;
      #pragma unroll
      for (int off = 1; off < 16; off <<= 1)
        mx = fmaxf(mx, __shfl_xor(mx, off, 32));
      float nm = fmaxf(mrow[r], mx);
      float corr = __expf(mrow[r] - nm);
      mrow[r] = nm;
      lrow[r] *= corr;
      #pragma unroll
      for (int nt = 0; nt < 4; ++nt) o[nt][r] *= corr;
      float rs = 0.0f;
      #pragma unroll
      for (int nt = 0; nt < 4; ++nt) {
        float pv = __expf(s[nt][r] * scale - nm);
        s[nt][r] = pv;
        rs += pv;
      }
      #pragma unroll
      for (int off = 1; off < 16; off <<= 1) rs += __shfl_xor(rs, off, 32);
      lrow[r] += rs;
      #pragma unroll
      for (int nt = 0; nt < 4; ++nt)
        pbuf[wave][lg * 8 + r][nt * 16 + ln] = (_Float16)s[nt][r];
    }
    // reload P as A-fragments (C-layout -> A-layout via LDS, wave-private)
    V16 pf[2];
    #pragma unroll
    for (int kc = 0; kc < 2; ++kc) {
      const _Float16* pp = &pbuf[wave][ln][kc * 32 + lg * 8];
      pf[kc].u[0] = *(const u4*)pp;
      pf[kc].u[1] = *(const u4*)(pp + 16);
    }
    // O += P * V
    #pragma unroll
    for (int dt = 0; dt < 4; ++dt) {
      const _Float16* vp = Vt + (size_t)(h * DH_ + dt * 16 + ln) * N_ + jb;
      #pragma unroll
      for (int kc = 0; kc < 2; ++kc) {
        V16 b;
        b.u[0] = *(const u4*)(vp + kc * 32 + lg * 8);
        b.u[1] = *(const u4*)(vp + kc * 32 + lg * 8 + 16);
        o[dt] = wmma16(pf[kc].v, b.v, o[dt]);
      }
    }
  }
  #pragma unroll
  for (int r = 0; r < 8; ++r) {
    float inv = 1.0f / lrow[r];
    int m = qrow + lg * 8 + r;
    #pragma unroll
    for (int dt = 0; dt < 4; ++dt)
      O[(size_t)m * D_ + h * DH_ + dt * 16 + ln] = (_Float16)(o[dt][r] * inv);
  }
}

// ---------------------------------------------------------------------------
// Fused residual + layernorm over D=512.  grid = N, block = 256.
__global__ __launch_bounds__(256) void ln_ws(const float* __restrict__ a,
                                             const float* __restrict__ b,
                                             const float* __restrict__ g,
                                             const float* __restrict__ be,
                                             float* __restrict__ outF,
                                             _Float16* __restrict__ outH) {
  const int tid = threadIdx.x;
  const size_t base = (size_t)blockIdx.x * D_;
  float x0 = a[base + tid]       + (b ? b[base + tid]       : 0.0f);
  float x1 = a[base + tid + 256] + (b ? b[base + tid + 256] : 0.0f);
  float mean = block_reduce(x0 + x1, 0) * (1.0f / D_);
  float d0 = x0 - mean, d1 = x1 - mean;
  float var = block_reduce(d0 * d0 + d1 * d1, 0) * (1.0f / D_);
  float rstd = rsqrtf(var + 1e-5f);
  float y0 = d0 * rstd * g[tid]       + be[tid];
  float y1 = d1 * rstd * g[tid + 256] + be[tid + 256];
  if (outF) { outF[base + tid] = y0; outF[base + tid + 256] = y1; }
  if (outH) { outH[base + tid] = (_Float16)y0; outH[base + tid + 256] = (_Float16)y1; }
}

// score[row] = hid[row,:] . w2 + b2   (HID=256).  grid = N, block = 256.
__global__ __launch_bounds__(256) void secscore_ws(const float* __restrict__ hid,
                                                   const float* __restrict__ w2,
                                                   const float* __restrict__ b2,
                                                   float* __restrict__ out) {
  int row = blockIdx.x;
  float v = hid[(size_t)row * HID_ + threadIdx.x] * w2[threadIdx.x];
  float s = block_reduce(v, 0);
  if (threadIdx.x == 0) out[row] = s + b2[0];
}

// Per-section masked softmax pool.  grid = S, block = 256.
__global__ __launch_bounds__(256) void secpool_ws(const long long* __restrict__ spans,
                                                  const float* __restrict__ score,
                                                  const float* __restrict__ nodes,
                                                  float* __restrict__ secv) {
  int s = blockIdx.x;
  int st = (int)spans[2 * s], en = (int)spans[2 * s + 1];
  int tid = threadIdx.x;
  float mx = -3.0e38f;
  for (int i = st + tid; i < en; i += 256) mx = fmaxf(mx, score[i]);
  mx = block_reduce(mx, 1);
  float se = 0.0f;
  for (int i = st + tid; i < en; i += 256) se += __expf(score[i] - mx);
  se = block_reduce(se, 0);
  float inv = 1.0f / se;
  for (int d = tid; d < D_; d += 256) {
    float acc = 0.0f;
    for (int i = st; i < en; ++i)
      acc += __expf(score[i] - mx) * nodes[(size_t)i * D_ + d];
    secv[(size_t)s * D_ + d] = acc * inv;
  }
}

// Document pool.  One block of 256 threads (HID == 256).
__global__ __launch_bounds__(256) void docpool_ws(const float* __restrict__ secv,
                                                  const float* __restrict__ dw1,
                                                  const float* __restrict__ db1,
                                                  const float* __restrict__ dw2,
                                                  const float* __restrict__ db2,
                                                  float* __restrict__ docv) {
  __shared__ float sw[S_];
  int tid = threadIdx.x;
  for (int s = 0; s < S_; ++s) {
    float p = 0.0f;
    for (int d = 0; d < D_; ++d)
      p += secv[(size_t)s * D_ + d] * dw1[(size_t)d * HID_ + tid];
    float hv = tanhf(p + db1[tid]) * dw2[tid];
    float sum = block_reduce(hv, 0);
    if (tid == 0) sw[s] = sum + db2[0];
  }
  __syncthreads();
  if (tid == 0) {
    float mx = -3.0e38f;
    for (int s = 0; s < S_; ++s) mx = fmaxf(mx, sw[s]);
    float se = 0.0f;
    for (int s = 0; s < S_; ++s) { sw[s] = __expf(sw[s] - mx); se += sw[s]; }
    float inv = 1.0f / se;
    for (int s = 0; s < S_; ++s) sw[s] *= inv;
  }
  __syncthreads();
  for (int d = tid; d < D_; d += 256) {
    float acc = 0.0f;
    for (int s = 0; s < S_; ++s) acc += sw[s] * secv[(size_t)s * D_ + d];
    docv[d] = acc;
  }
}

// ---------------------------------------------------------------------------
extern "C" void kernel_launch(void* const* d_in, const int* in_sizes, int n_in,
                              void* d_out, int out_size, void* d_ws, size_t ws_size,
                              hipStream_t stream) {
  (void)in_sizes; (void)n_in; (void)out_size; (void)ws_size;
  const float*     x     = (const float*)d_in[0];
  const long long* spans = (const long long*)d_in[1];
  const float* wq = (const float*)d_in[2];   const float* bq = (const float*)d_in[3];
  const float* wk = (const float*)d_in[4];   const float* bk = (const float*)d_in[5];
  const float* wv = (const float*)d_in[6];   const float* bv = (const float*)d_in[7];
  const float* wo = (const float*)d_in[8];   const float* bo = (const float*)d_in[9];
  const float* ln1g = (const float*)d_in[10]; const float* ln1b = (const float*)d_in[11];
  const float* ln2g = (const float*)d_in[12]; const float* ln2b = (const float*)d_in[13];
  const float* w1 = (const float*)d_in[14];  const float* b1 = (const float*)d_in[15];
  const float* w2 = (const float*)d_in[16];  const float* b2 = (const float*)d_in[17];
  const float* sw1 = (const float*)d_in[18]; const float* sb1 = (const float*)d_in[19];
  const float* sw2 = (const float*)d_in[20]; const float* sb2 = (const float*)d_in[21];
  const float* dw1 = (const float*)d_in[22]; const float* db1 = (const float*)d_in[23];
  const float* dw2 = (const float*)d_in[24]; const float* db2 = (const float*)d_in[25];

  char* wp = (char*)d_ws;
  auto alloc = [&](size_t bytes) -> void* {
    void* r = (void*)wp;
    wp += (bytes + 255) & ~(size_t)255;
    return r;
  };
  _Float16* xb   = (_Float16*)alloc((size_t)N_ * D_ * 2);
  _Float16* wqT  = (_Float16*)alloc((size_t)D_ * D_ * 2);
  _Float16* wkT  = (_Float16*)alloc((size_t)D_ * D_ * 2);
  _Float16* wvT  = (_Float16*)alloc((size_t)D_ * D_ * 2);
  _Float16* woT  = (_Float16*)alloc((size_t)D_ * D_ * 2);
  _Float16* w1T  = (_Float16*)alloc((size_t)D_ * 2 * D_ * 2);
  _Float16* w2T  = (_Float16*)alloc((size_t)D_ * 2 * D_ * 2);
  _Float16* sw1T = (_Float16*)alloc((size_t)HID_ * D_ * 2);
  _Float16* qb   = (_Float16*)alloc((size_t)N_ * D_ * 2);
  _Float16* kb   = (_Float16*)alloc((size_t)N_ * D_ * 2);
  _Float16* vb   = (_Float16*)alloc((size_t)N_ * D_ * 2);
  _Float16* vT   = (_Float16*)alloc((size_t)N_ * D_ * 2);
  _Float16* ob   = (_Float16*)alloc((size_t)N_ * D_ * 2);
  float*    aproj= (float*)   alloc((size_t)N_ * D_ * 4);
  float*    x1   = (float*)   alloc((size_t)N_ * D_ * 4);
  _Float16* x1b  = (_Float16*)alloc((size_t)N_ * D_ * 2);
  _Float16* h1b  = (_Float16*)alloc((size_t)N_ * 2 * D_ * 2);
  float*    ffn2 = (float*)   alloc((size_t)N_ * D_ * 4);
  _Float16* nodb = (_Float16*)alloc((size_t)N_ * D_ * 2);
  float*    sech = (float*)   alloc((size_t)N_ * HID_ * 4);
  float*    sscr = (float*)   alloc((size_t)N_ * 4);

  float* nodes = (float*)d_out;
  float* secv  = nodes + (size_t)N_ * D_;
  float* docv  = secv + (size_t)S_ * D_;

  // ---- prep: fp32 -> fp16 conversions and weight transposes
  cvt_f32_f16<<<(N_ * D_ + 255) / 256, 256, 0, stream>>>(x, xb, N_ * D_);
  trans_f32_f16<<<(D_ * D_ + 255) / 256, 256, 0, stream>>>(wq, wqT, D_, D_);
  trans_f32_f16<<<(D_ * D_ + 255) / 256, 256, 0, stream>>>(wk, wkT, D_, D_);
  trans_f32_f16<<<(D_ * D_ + 255) / 256, 256, 0, stream>>>(wv, wvT, D_, D_);
  trans_f32_f16<<<(D_ * D_ + 255) / 256, 256, 0, stream>>>(wo, woT, D_, D_);
  trans_f32_f16<<<(D_ * 2 * D_ + 255) / 256, 256, 0, stream>>>(w1, w1T, D_, 2 * D_);
  trans_f32_f16<<<(D_ * 2 * D_ + 255) / 256, 256, 0, stream>>>(w2, w2T, 2 * D_, D_);
  trans_f32_f16<<<(D_ * HID_ + 255) / 256, 256, 0, stream>>>(sw1, sw1T, D_, HID_);

  // ---- QKV projections (WMMA)
  dim3 blk(256), gqkv(D_ / 64, N_ / 128);
  gemm_ws<<<gqkv, blk, 0, stream>>>(xb, wqT, bq, nullptr, qb, N_, D_, D_, 0);
  gemm_ws<<<gqkv, blk, 0, stream>>>(xb, wkT, bk, nullptr, kb, N_, D_, D_, 0);
  gemm_ws<<<gqkv, blk, 0, stream>>>(xb, wvT, bv, nullptr, vb, N_, D_, D_, 0);
  trans_f16<<<(N_ * D_ + 255) / 256, 256, 0, stream>>>(vb, vT, N_, D_);

  // ---- flash attention (WMMA)
  fa_ws<<<dim3(N_ / 64, HEADS_), dim3(128), 0, stream>>>(qb, kb, vT, ob);

  // ---- output projection + residual + LN1
  gemm_ws<<<gqkv, blk, 0, stream>>>(ob, woT, bo, aproj, nullptr, N_, D_, D_, 0);
  ln_ws<<<N_, 256, 0, stream>>>(x, aproj, ln1g, ln1b, x1, x1b);

  // ---- FFN (WMMA, GELU fused) + residual + LN2 -> nodes (output 0)
  gemm_ws<<<dim3(2 * D_ / 64, N_ / 128), blk, 0, stream>>>(x1b, w1T, b1, nullptr, h1b,
                                                           N_, 2 * D_, D_, 1);
  gemm_ws<<<gqkv, blk, 0, stream>>>(h1b, w2T, b2, ffn2, nullptr, N_, D_, 2 * D_, 0);
  ln_ws<<<N_, 256, 0, stream>>>(x1, ffn2, ln2g, ln2b, nodes, nodb);

  // ---- section pooling (WMMA hidden, tanh fused) -> section_vecs (output 1)
  gemm_ws<<<dim3(HID_ / 64, N_ / 128), blk, 0, stream>>>(nodb, sw1T, sb1, sech, nullptr,
                                                         N_, HID_, D_, 2);
  secscore_ws<<<N_, 256, 0, stream>>>(sech, sw2, sb2, sscr);
  secpool_ws<<<S_, 256, 0, stream>>>(spans, sscr, nodes, secv);

  // ---- document pooling -> document_vec (output 2)
  docpool_ws<<<1, 256, 0, stream>>>(secv, dw1, db1, dw2, db2, docv);
}
